// CapsuleNetwork_45466523796148
// MI455X (gfx1250) — matrix operand order repaired and checked by reading, not compiled, not run on hardware
//
#include <hip/hip_runtime.h>
#include <hip/hip_bf16.h>

// Problem constants (match reference)
#define Bn 512
#define Sn 128
#define Kn 4
#define Hn 128
#define KH (Kn * Hn)   // 512
#define EPSF 1e-9f

typedef __bf16 v16bf __attribute__((ext_vector_type(16)));
typedef __bf16 v8bf  __attribute__((ext_vector_type(8)));
typedef float  v8f   __attribute__((ext_vector_type(8)));

static __device__ __forceinline__ __bf16 f2bf(float f) {
    unsigned u = __builtin_bit_cast(unsigned, f);
    u += 0x7FFFu + ((u >> 16) & 1u);           // round-to-nearest-even
    unsigned short h = (unsigned short)(u >> 16);
    return __builtin_bit_cast(__bf16, h);
}

static __device__ __forceinline__ v16bf cat16(v8bf lo, v8bf hi) {
    return __builtin_shufflevector(lo, hi, 0,1,2,3,4,5,6,7,8,9,10,11,12,13,14,15);
}

// ---------------------------------------------------------------------------
// Stage 1: item_eb_hat[b,k,s,h] = sum_h' item_eb[b,s,h'] * w[s, k*H+h, h']
// Per-s GEMM: A = item_eb[:,s,:]  (512x128, K-major rows)
//             B = w[s,:,:]^T      (128x512; w rows are K-major -> B columns)
// grid = (s=128, mb=4, nb=4), block = 256 threads = 8 waves.
// Block computes a 128(M) x 128(N) tile; nb is exactly the capsule index k.
// ---------------------------------------------------------------------------
__global__ __launch_bounds__(256) void bilinear_wmma_kernel(
        const float* __restrict__ item_eb,   // [B,S,H]
        const float* __restrict__ w,         // [S,KH,H]
        float* __restrict__ hat)             // [B,K,S,H]
{
    __shared__ __align__(16) __bf16 As[128][136];  // [m][k]  (+8 pad: bank rotate)
    __shared__ __align__(16) __bf16 Ws[128][136];  // [n][k]

    const int s   = blockIdx.x;
    const int mb  = blockIdx.y;
    const int nb  = blockIdx.z;
    const int tid = threadIdx.x;

    // ---- stage both 128x128 f32 panels into LDS as bf16 ----
    #pragma unroll
    for (int it = 0; it < 16; ++it) {
        const int f   = it * 256 + tid;        // 0..4095 float4 chunks
        const int row = f >> 5;                // 0..127
        const int c4  = (f & 31) << 2;         // 0..124 step 4

        const float4 a = *(const float4*)(item_eb +
                ((size_t)(mb * 128 + row) * Sn + s) * Hn + c4);
        As[row][c4 + 0] = f2bf(a.x);
        As[row][c4 + 1] = f2bf(a.y);
        As[row][c4 + 2] = f2bf(a.z);
        As[row][c4 + 3] = f2bf(a.w);

        const float4 b = *(const float4*)(w +
                ((size_t)s * KH + nb * 128 + row) * Hn + c4);
        Ws[row][c4 + 0] = f2bf(b.x);
        Ws[row][c4 + 1] = f2bf(b.y);
        Ws[row][c4 + 2] = f2bf(b.z);
        Ws[row][c4 + 3] = f2bf(b.w);
    }
    __syncthreads();

    const int lane = tid & 31;
    const int wid  = tid >> 5;        // 0..7
    const int wm   = wid & 3;         // 0..3 -> 32 M rows each
    const int wn   = wid >> 2;        // 0..1 -> 64 N cols each
    const int l16  = lane & 15;
    const int hiH  = lane >> 4;       // 0 / 1

    v8f acc[2][4];
    #pragma unroll
    for (int tm = 0; tm < 2; ++tm)
        #pragma unroll
        for (int tn = 0; tn < 4; ++tn)
            acc[tm][tn] = (v8f)(0.0f);

    #pragma unroll
    for (int ks = 0; ks < 4; ++ks) {
        const int k0  = ks * 32;
        // A fragment (16-bit 16x32): lane<16 -> K {0..7,16..23}; lane>=16 -> +8
        const int kbA = k0 + (hiH ? 8 : 0);
        v16bf afrag[2];
        #pragma unroll
        for (int tm = 0; tm < 2; ++tm) {
            const int m = wm * 32 + tm * 16 + l16;
            v8bf lo = *(const v8bf*)&As[m][kbA];
            v8bf hi = *(const v8bf*)&As[m][kbA + 16];
            afrag[tm] = cat16(lo, hi);
        }
        // B fragment (16-bit 32x16): lane<16 -> K {0..7,16..23}; lane>=16 -> +8
        const int kbB = k0 + (hiH ? 8 : 0);
        v16bf bfrag[4];
        #pragma unroll
        for (int tn = 0; tn < 4; ++tn) {
            const int n = wn * 64 + tn * 16 + l16;
            v8bf lo = *(const v8bf*)&Ws[n][kbB];
            v8bf hi = *(const v8bf*)&Ws[n][kbB + 16];
            bfrag[tn] = cat16(lo, hi);
        }
        #pragma unroll
        for (int tm = 0; tm < 2; ++tm)
            #pragma unroll
            for (int tn = 0; tn < 4; ++tn)
                acc[tm][tn] = __builtin_amdgcn_wmma_f32_16x16x32_bf16(
                        false, afrag[tm], false, bfrag[tn],
                        (short)0, acc[tm][tn], false, false);
    }

    // ---- store C: D layout VGPR r: lanes0-15 -> M=r, lanes16-31 -> M=r+8 ----
    #pragma unroll
    for (int tm = 0; tm < 2; ++tm) {
        #pragma unroll
        for (int tn = 0; tn < 4; ++tn) {
            const int hcol = wn * 64 + tn * 16 + l16;        // h in [0,128)
            #pragma unroll
            for (int r = 0; r < 8; ++r) {
                const int mloc = wm * 32 + tm * 16 + r + (hiH ? 8 : 0);
                const int bg   = mb * 128 + mloc;
                hat[(((size_t)bg * Kn + nb) * Sn + s) * Hn + hcol] = acc[tm][tn][r];
            }
        }
    }
}

// ---------------------------------------------------------------------------
// Zero-fill capsule_weight
// ---------------------------------------------------------------------------
__global__ void zero_f32_kernel(float* __restrict__ p, int n) {
    const int i = blockIdx.x * 256 + threadIdx.x;
    if (i < n) p[i] = 0.0f;
}

// ---------------------------------------------------------------------------
// Softmax over axis 0 (batch): per column c = k*S+s, reduce over b=0..511.
// grid = 512 blocks of 256 threads.
// ---------------------------------------------------------------------------
__global__ __launch_bounds__(256) void softmax_stats_kernel(
        const float* __restrict__ cw,   // [B, K*S]  (b-major)
        float* __restrict__ smax,       // [K*S]
        float* __restrict__ sinv)       // [K*S]
{
    __shared__ float red[256];
    const int c = blockIdx.x;
    const int t = threadIdx.x;

    const float a0 = cw[(size_t)t * (Kn * Sn) + c];
    const float a1 = cw[(size_t)(t + 256) * (Kn * Sn) + c];

    red[t] = fmaxf(a0, a1);
    __syncthreads();
    for (int off = 128; off; off >>= 1) {
        if (t < off) red[t] = fmaxf(red[t], red[t + off]);
        __syncthreads();
    }
    const float mx = red[0];
    __syncthreads();

    red[t] = __expf(a0 - mx) + __expf(a1 - mx);
    __syncthreads();
    for (int off = 128; off; off >>= 1) {
        if (t < off) red[t] += red[t + off];
        __syncthreads();
    }
    if (t == 0) { smax[c] = mx; sinv[c] = 1.0f / red[0]; }
}

// ---------------------------------------------------------------------------
// One routing step for one (b,k): v = sw . hat; squash; delta / final store.
// grid = B*K = 2048 blocks of 128 threads (t = h or t = s).
// ---------------------------------------------------------------------------
__global__ __launch_bounds__(128) void routing_step_kernel(
        const float* __restrict__ hat,   // [B,K,S,H]
        const float* __restrict__ mask,  // [B,S]
        float* __restrict__ cw,          // [B, K*S]
        const float* __restrict__ smax,
        const float* __restrict__ sinv,
        float* __restrict__ out,         // [B,K,H]
        int last)
{
    __shared__ float swrow[Sn];
    __shared__ float caprow[Hn];
    __shared__ float red[Hn];

    const int bk = blockIdx.x;
    const int b  = bk >> 2;
    const int k  = bk & 3;
    const int t  = threadIdx.x;

    {   // masked softmax weight for s = t
        const int c = k * Sn + t;
        const float sw = __expf(cw[(size_t)b * (Kn * Sn) + c] - smax[c]) * sinv[c];
        swrow[t] = (mask[(size_t)b * Sn + t] == 0.0f) ? 0.0f : sw;
    }
    __syncthreads();

    const float* hrow = hat + (size_t)bk * Sn * Hn;   // [s][h]
    __builtin_prefetch(hrow + (size_t)t * Hn, 0, 1);  // global_prefetch_b8

    // v[h=t] = sum_s sw[s] * hat[s][t]
    float v = 0.0f;
    #pragma unroll 4
    for (int s2 = 0; s2 < Sn; ++s2) v = fmaf(swrow[s2], hrow[s2 * Hn + t], v);

    // squash: n = |v|^2 over H
    red[t] = v * v;
    __syncthreads();
    for (int off = 64; off; off >>= 1) {
        if (t < off) red[t] += red[t + off];
        __syncthreads();
    }
    const float n = red[0];
    const float scale = (n / (1.0f + n)) * rsqrtf(n + EPSF);
    const float cap = scale * v;
    caprow[t] = cap;
    __syncthreads();

    if (last) {
        out[(size_t)bk * Hn + t] = cap;
    } else {
        // delta[s=t] = sum_h hat[t][h] * cap[h]; cw += delta
        const float* hr = hrow + (size_t)t * Hn;
        float d = 0.0f;
        #pragma unroll 4
        for (int h2 = 0; h2 < Hn; ++h2) d = fmaf(hr[h2], caprow[h2], d);
        cw[(size_t)b * (Kn * Sn) + k * Sn + t] += d;
    }
}

// ---------------------------------------------------------------------------
extern "C" void kernel_launch(void* const* d_in, const int* in_sizes, int n_in,
                              void* d_out, int out_size, void* d_ws, size_t ws_size,
                              hipStream_t stream) {
    (void)in_sizes; (void)n_in; (void)out_size; (void)ws_size;

    const float* item_eb = (const float*)d_in[0];   // [B,S,H]
    const float* mask    = (const float*)d_in[1];   // [B,S]
    const float* w       = (const float*)d_in[2];   // [1,S,KH,H]
    float*       out     = (float*)d_out;           // [B,K,H]

    // workspace layout
    float* hat  = (float*)d_ws;                               // B*K*S*H
    float* cw   = hat  + (size_t)Bn * Kn * Sn * Hn;           // B*K*S
    float* smax = cw   + (size_t)Bn * Kn * Sn;                // K*S
    float* sinv = smax + (size_t)Kn * Sn;                     // K*S

    // Stage 1: bilinear map via bf16 WMMA
    dim3 g1(Sn, 4, 4);
    bilinear_wmma_kernel<<<g1, 256, 0, stream>>>(item_eb, w, hat);

    // capsule_weight = 0
    const int ncw = Bn * Kn * Sn;
    zero_f32_kernel<<<(ncw + 255) / 256, 256, 0, stream>>>(cw, ncw);

    // Stage 2: 3 routing iterations (hat_iter == item_eb_hat numerically)
    for (int i = 0; i < 3; ++i) {
        softmax_stats_kernel<<<Kn * Sn, 256, 0, stream>>>(cw, smax, sinv);
        routing_step_kernel<<<Bn * Kn, 128, 0, stream>>>(hat, mask, cw, smax, sinv,
                                                         out, (i == 2) ? 1 : 0);
    }
}
